// HiddenMarkovModel_23338852286946
// MI455X (gfx1250) — matrix operand order repaired and checked by reading, not compiled, not run on hardware
//
#include <hip/hip_runtime.h>
#include <hip/hip_bf16.h>

// Problem constants (match reference)
#define KTAG 128
#define VOC  50000
#define DIM  512
#define BATCH 256
#define NLEN 512
#define BOS_T (KTAG - 1)
#define EOS_T (KTAG - 2)
#define NEGBIG (-1e30f)
#define CHUNK 196          // ceil(VOC/256)
#define VPAD  50048        // 782 * 64 word rows in WBt

typedef __attribute__((ext_vector_type(16))) __bf16 v16bf;
typedef __attribute__((ext_vector_type(8)))  float  v8f;

union FragAB { v16bf v; unsigned u[8]; };

__device__ __forceinline__ unsigned bf16rne(float x) {
    unsigned u = __float_as_uint(x);
    u = (u + 0x7FFFu + ((u >> 16) & 1u)) >> 16;   // round-to-nearest-even
    return u;
}
__device__ __forceinline__ unsigned bf16pair(float a, float b) {
    return bf16rne(a) | (bf16rne(b) << 16);
}

// ---------------------------------------------------------------------------
// Kernel 1: A = softmax(WA with BOS column masked) = exp(log_softmax(WA_masked))
// ---------------------------------------------------------------------------
__global__ __launch_bounds__(KTAG) void k_transmat(const float* __restrict__ WA,
                                                   float* __restrict__ Aexp) {
    const int i = threadIdx.x;               // row (prev tag)
    float m = -3.4e38f;
    for (int j = 0; j < KTAG; ++j) {
        float w = (j == BOS_T) ? NEGBIG : WA[i * KTAG + j];
        m = fmaxf(m, w);
    }
    float s = 0.f;
    for (int j = 0; j < KTAG; ++j) {
        float w = (j == BOS_T) ? NEGBIG : WA[i * KTAG + j];
        s += __expf(w - m);
    }
    const float inv = 1.f / s;
    for (int j = 0; j < KTAG; ++j) {
        float w = (j == BOS_T) ? NEGBIG : WA[i * KTAG + j];
        Aexp[i * KTAG + j] = __expf(w - m) * inv;
    }
}

// ---------------------------------------------------------------------------
// Kernel 2: WBt[word][tag] = E @ ThetaB^T  via v_wmma_f32_16x16x32_bf16
//   Block: 256 threads (8 waves). Tile: 64 words (M) x 128 tags (N).
//   Wave w -> tag subtile w*16; 4 word subtiles; K=DIM staged in LDS chunks of 64.
// ---------------------------------------------------------------------------
__global__ __launch_bounds__(256) void k_gemm(const float* __restrict__ ThetaB,
                                              const float* __restrict__ E,
                                              float* __restrict__ WBt) {
    __shared__ unsigned Tlds[KTAG * 36];     // 128 tag rows x 32 u32 (64 bf16 d) + pad
    __shared__ unsigned Elds[64 * 36];       // 64 word rows x 32 u32 + pad

    const int tid  = threadIdx.x;
    const int wave = tid >> 5;               // 0..7 -> tag tile
    const int lane = tid & 31;
    const int lhi  = lane >> 4;              // half-wave select
    const int llo  = lane & 15;
    const int wbase = blockIdx.x * 64;       // first word of this block

    v8f acc[4];
    #pragma unroll
    for (int m = 0; m < 4; ++m) acc[m] = (v8f){0.f,0.f,0.f,0.f,0.f,0.f,0.f,0.f};

    const int rquo = tid >> 3;               // 0..31
    const int rp   = tid & 7;                // 8 threads per row, 8 floats each

    for (int d0 = 0; d0 < DIM; d0 += 64) {
        __syncthreads();
        // stage ThetaB chunk: 128 rows x 64 d (f32 -> packed bf16), b128 loads
        #pragma unroll
        for (int pass = 0; pass < 4; ++pass) {
            const int r = pass * 32 + rquo;
            const float4* src = (const float4*)(ThetaB + (size_t)r * DIM + d0 + rp * 8);
            const float4 x0 = src[0], x1 = src[1];
            unsigned* dst = &Tlds[r * 36 + rp * 4];
            dst[0] = bf16pair(x0.x, x0.y);
            dst[1] = bf16pair(x0.z, x0.w);
            dst[2] = bf16pair(x1.x, x1.y);
            dst[3] = bf16pair(x1.z, x1.w);
        }
        // stage E chunk: 64 word rows x 64 d (clamped at tail) + prefetch next chunk
        #pragma unroll
        for (int pass = 0; pass < 2; ++pass) {
            const int r = pass * 32 + rquo;
            int w = wbase + r; if (w > VOC - 1) w = VOC - 1;
            const float* rowp = E + (size_t)w * DIM + d0 + rp * 8;
            if (d0 + 64 < DIM) __builtin_prefetch(rowp + 64, 0, 1);  // next E chunk -> L2/L0
            const float4* src = (const float4*)rowp;
            const float4 x0 = src[0], x1 = src[1];
            unsigned* dst = &Elds[r * 36 + rp * 4];
            dst[0] = bf16pair(x0.x, x0.y);
            dst[1] = bf16pair(x0.z, x0.w);
            dst[2] = bf16pair(x1.x, x1.y);
            dst[3] = bf16pair(x1.z, x1.w);
        }
        __syncthreads();

        #pragma unroll
        for (int dk2 = 0; dk2 < 32; dk2 += 16) {     // K=32 bf16 per wmma (16 u32)
            // B fragment: ThetaB^T tile [32 d x 16 tags]; lane: N=llo, K=lhi*16 + 2v+h
            FragAB fb;
            {
                const unsigned* bp = &Tlds[(wave * 16 + llo) * 36 + dk2 + lhi * 8];
                #pragma unroll
                for (int q = 0; q < 8; ++q) fb.u[q] = bp[q];
            }
            // Preload ALL 4 A fragments, then issue 4 back-to-back WMMAs
            FragAB fa[4];
            #pragma unroll
            for (int msub = 0; msub < 4; ++msub) {
                const unsigned* ap = &Elds[(msub * 16 + llo) * 36 + dk2 + lhi * 8];
                #pragma unroll
                for (int q = 0; q < 4; ++q) { fa[msub].u[q] = ap[q]; fa[msub].u[4 + q] = ap[4 + q]; }
            }
            #pragma unroll
            for (int msub = 0; msub < 4; ++msub) {
                acc[msub] = __builtin_amdgcn_wmma_f32_16x16x32_bf16(
                    false, fa[msub].v, false, fb.v, (short)0, acc[msub], false, false);
            }
        }
    }

    // Store D tiles: lane l, vgpr v -> word = base + msub*16 + v + 8*lhi, tag = wave*16+llo
    const int tag = wave * 16 + llo;
    if (wbase + 64 <= VOC) {                 // uniform fast path (781 of 782 blocks)
        #pragma unroll
        for (int msub = 0; msub < 4; ++msub) {
            #pragma unroll
            for (int v = 0; v < 8; ++v) {
                const int w = wbase + msub * 16 + v + 8 * lhi;
                WBt[(size_t)w * KTAG + tag] = acc[msub][v];
            }
        }
    } else {
        #pragma unroll
        for (int msub = 0; msub < 4; ++msub) {
            #pragma unroll
            for (int v = 0; v < 8; ++v) {
                const int w = wbase + msub * 16 + v + 8 * lhi;
                if (w < VOC) WBt[(size_t)w * KTAG + tag] = acc[msub][v];
            }
        }
    }
}

// ---------------------------------------------------------------------------
// Kernel 3: per-chunk online logsumexp over the vocab axis (rows of WB)
// ---------------------------------------------------------------------------
__global__ __launch_bounds__(KTAG) void k_lse_part(const float* __restrict__ WBt,
                                                   float* __restrict__ pm,
                                                   float* __restrict__ ps) {
    const int t = threadIdx.x, b = blockIdx.x;
    const int v0 = b * CHUNK;
    int v1 = v0 + CHUNK; if (v1 > VOC) v1 = VOC;
    float m = -3.4e38f, s = 0.f;
    for (int v = v0; v < v1; ++v) {
        const float x = WBt[(size_t)v * KTAG + t];
        const float nm = fmaxf(m, x);
        s = s * __expf(m - nm) + __expf(x - nm);
        m = nm;
    }
    pm[b * KTAG + t] = m;
    ps[b * KTAG + t] = s;
}

__global__ __launch_bounds__(KTAG) void k_lse_comb(const float* __restrict__ pm,
                                                   const float* __restrict__ ps,
                                                   float* __restrict__ logden) {
    const int t = threadIdx.x;
    float M = -3.4e38f, S = 0.f;
    for (int c = 0; c < 256; ++c) {
        const float m = pm[c * KTAG + t], s = ps[c * KTAG + t];
        const float nM = fmaxf(M, m);
        S = S * __expf(M - nM) + s * __expf(m - nM);
        M = nM;
    }
    logden[t] = M + __logf(S);
}

// ---------------------------------------------------------------------------
// Kernel 4: forward scan. One block = one sentence; thread j = tag j.
//   alpha'[j] = m + log( sum_i exp(alpha[i]-m) * A[i][j] ) + emiss[j]
//   A in dynamic LDS (64 KB); token emissions gathered from L2-resident WBt.
// ---------------------------------------------------------------------------
__global__ __launch_bounds__(KTAG) void k_scan(const float* __restrict__ Aexp,
                                               const float* __restrict__ WBt,
                                               const float* __restrict__ logden,
                                               const int*   __restrict__ words,
                                               float* __restrict__ Z) {
    extern __shared__ float Alds[];              // KTAG*KTAG floats (64 KB)
    __shared__ __align__(16) float plds[KTAG];
    __shared__ float red[4];
    __shared__ int   wlds[NLEN];

    const int j = threadIdx.x, b = blockIdx.x;
    const int lane = j & 31, wv = j >> 5;

    // vectorized fill of the transition table (b128 loads/stores)
    {
        float4* A4 = (float4*)Alds;
        const float4* G4 = (const float4*)Aexp;
        #pragma unroll 4
        for (int q = 0; q < (KTAG * KTAG) / (4 * KTAG); ++q)   // 32 iters
            A4[q * KTAG + j] = G4[q * KTAG + j];
    }
    #pragma unroll
    for (int q = 0; q < NLEN / KTAG; ++q)
        wlds[q * KTAG + j] = words[(size_t)b * NLEN + q * KTAG + j];

    const float ld = logden[j];
    const bool dead = (j == BOS_T) || (j == EOS_T);  // tags that emit nothing
    float aj = (j == BOS_T) ? 1.0f : NEGBIG;          // reference's alpha0 quirk
    __syncthreads();

    for (int n = 0; n < NLEN; ++n) {
        // block max of alpha: wave shuffles + 4-entry LDS combine
        float m = aj;
        m = fmaxf(m, __shfl_xor(m, 16));
        m = fmaxf(m, __shfl_xor(m, 8));
        m = fmaxf(m, __shfl_xor(m, 4));
        m = fmaxf(m, __shfl_xor(m, 2));
        m = fmaxf(m, __shfl_xor(m, 1));
        if (lane == 0) red[wv] = m;
        __syncthreads();
        m = fmaxf(fmaxf(red[0], red[1]), fmaxf(red[2], red[3]));
        plds[j] = __expf(aj - m);
        __syncthreads();

        float s = 0.f;
        #pragma unroll 4
        for (int i = 0; i < KTAG; i += 4) {
            const float4 p4 = *(const float4*)&plds[i];      // one b128 broadcast
            s = fmaf(p4.x, Alds[(i + 0) * KTAG + j], s);     // conflict-free taps
            s = fmaf(p4.y, Alds[(i + 1) * KTAG + j], s);
            s = fmaf(p4.z, Alds[(i + 2) * KTAG + j], s);
            s = fmaf(p4.w, Alds[(i + 3) * KTAG + j], s);
        }

        const int w = wlds[n];
        const float e = dead ? NEGBIG : (WBt[(size_t)w * KTAG + j] - ld);
        aj = m + __logf(fmaxf(s, 1e-37f)) + e;
        __syncthreads();                                 // protect plds/red rewrite
    }

    // Final EOS transition: Z = m + log( sum_j exp(aj - m) * A[j][EOS] )
    float m = aj;
    m = fmaxf(m, __shfl_xor(m, 16));
    m = fmaxf(m, __shfl_xor(m, 8));
    m = fmaxf(m, __shfl_xor(m, 4));
    m = fmaxf(m, __shfl_xor(m, 2));
    m = fmaxf(m, __shfl_xor(m, 1));
    if (lane == 0) red[wv] = m;
    __syncthreads();
    m = fmaxf(fmaxf(red[0], red[1]), fmaxf(red[2], red[3]));
    __syncthreads();

    float t = __expf(aj - m) * Alds[j * KTAG + EOS_T];
    t += __shfl_xor(t, 16);
    t += __shfl_xor(t, 8);
    t += __shfl_xor(t, 4);
    t += __shfl_xor(t, 2);
    t += __shfl_xor(t, 1);
    if (lane == 0) red[wv] = t;
    __syncthreads();
    if (j == 0) Z[b] = m + __logf(red[0] + red[1] + red[2] + red[3]);
}

// ---------------------------------------------------------------------------
extern "C" void kernel_launch(void* const* d_in, const int* in_sizes, int n_in,
                              void* d_out, int out_size, void* d_ws, size_t ws_size,
                              hipStream_t stream) {
    const float* ThetaB = (const float*)d_in[0];
    const float* WA     = (const float*)d_in[1];
    const float* E      = (const float*)d_in[2];
    const int*   words  = (const int*)d_in[3];
    float* out = (float*)d_out;

    char* ws = (char*)d_ws;
    size_t off = 0;
    float* WBt    = (float*)(ws + off); off += (size_t)VPAD * KTAG * sizeof(float);
    float* Aexp   = (float*)(ws + off); off += (size_t)KTAG * KTAG * sizeof(float);
    float* pm     = (float*)(ws + off); off += (size_t)256 * KTAG * sizeof(float);
    float* ps     = (float*)(ws + off); off += (size_t)256 * KTAG * sizeof(float);
    float* logden = (float*)(ws + off); off += (size_t)KTAG * sizeof(float);

    k_transmat<<<1, KTAG, 0, stream>>>(WA, Aexp);
    k_gemm<<<(VOC + 63) / 64, 256, 0, stream>>>(ThetaB, E, WBt);
    k_lse_part<<<256, KTAG, 0, stream>>>(WBt, pm, ps);
    k_lse_comb<<<1, KTAG, 0, stream>>>(pm, ps, logden);
    k_scan<<<BATCH, KTAG, KTAG * KTAG * sizeof(float), stream>>>(Aexp, WBt, logden, words, out);
}
